// CustomLSTM_704374636591
// MI455X (gfx1250) — matrix operand order repaired and checked by reading, not compiled, not run on hardware
//
#include <hip/hip_runtime.h>
#include <hip/hip_bf16.h>

// ---------------------------------------------------------------------------
// Persistent fused LSTM for MI455X (gfx1250, wave32, WMMA bf16 16x16x32).
// B=32, T=2048, I=H=512.  64 persistent workgroups; each owns 8 hidden units
// (= 32 gate columns: i/f/c/o x 8).  Weights pinned in LDS as bf16, cell
// state c in registers, h exchanged via double-buffered global bf16 buffer
// with an atomics grid barrier each timestep.  x-GEMM and h-GEMM run as two
// interleaved independent WMMA accumulation chains to hide WMMA latency;
// activations use branch-free exp-based sigmoid/tanh on the critical path.
// ---------------------------------------------------------------------------

typedef __attribute__((ext_vector_type(16))) __bf16   v16bf;
typedef __attribute__((ext_vector_type(8)))  float    v8f;
typedef __attribute__((ext_vector_type(4)))  unsigned u32x4;
typedef __attribute__((ext_vector_type(2)))  float    f32x2;

#define B_   32
#define T_   2048
#define I_   512
#define H_   512
#define NWG  64        // persistent workgroups
#define HJ   8         // hidden units per WG
#define GC   32        // gate columns per WG (4*HJ)
#define SW   260       // dword stride of a K=512 bf16 row (520 bf16, 8 pad)
#define GSTR 36        // gates row stride in floats (bank-conflict pad)
#define NTHR 128       // 4 waves

__device__ __forceinline__ unsigned pk_bf16(float a, float b) {
  unsigned ua = __float_as_uint(a), ub = __float_as_uint(b);
  ua = (ua + 0x7FFFu + ((ua >> 16) & 1u)) >> 16;   // RNE f32 -> bf16
  ub = (ub + 0x7FFFu + ((ub >> 16) & 1u)) >> 16;
  return (ua & 0xFFFFu) | (ub << 16);
}

__device__ __forceinline__ float sigm(float v) { return 1.0f / (1.0f + __expf(-v)); }
// branch-free tanh: tanh(x) = 2*sigmoid(2x) - 1   (v_exp_f32 + v_rcp_f32 path)
__device__ __forceinline__ float tanh_fast(float v) {
  return __builtin_fmaf(2.0f, sigm(2.0f * v), -1.0f);
}

__global__ void lstm_init_kernel(unsigned* sync) {
  if (threadIdx.x == 0) { sync[0] = 0u; sync[1] = 0u; }
}

__global__ __launch_bounds__(NTHR) void lstm_persist_kernel(
    const float* __restrict__ x,  const float* __restrict__ Wi,
    const float* __restrict__ bi, const float* __restrict__ Wh,
    const float* __restrict__ bh, float* __restrict__ out,
    unsigned* __restrict__ sync,  unsigned short* __restrict__ hbuf)
{
  __shared__ unsigned sWi[GC * SW];    // bf16 Wi slice, rows = local gate col
  __shared__ unsigned sWh[GC * SW];    // bf16 Wh slice
  __shared__ unsigned sX [B_ * SW];    // bf16 x_t tile [32][512]
  __shared__ unsigned sH [B_ * SW];    // bf16 h_{t-1}  [32][512]
  __shared__ float    gates[B_ * GSTR];
  __shared__ float    bsum[GC];

  const int tid = threadIdx.x;
  const int wg  = blockIdx.x;
  const int j0  = wg * HJ;

  // ---- stage weight slices into LDS (once) -------------------------------
  {
    int r  = tid >> 2;            // local gate column 0..31
    int c0 = (tid & 3) * 64;      // dword chunk base (64 dwords = 128 floats)
    int g  = (r >> 3) * H_ + j0 + (r & 7);   // global gate row
    const float* wi = Wi + (size_t)g * I_ + (size_t)c0 * 2;
    const float* wh = Wh + (size_t)g * H_ + (size_t)c0 * 2;
    unsigned* di = &sWi[r * SW + c0];
    unsigned* dh = &sWh[r * SW + c0];
    for (int q = 0; q < 64; ++q) {
      f32x2 a = *(const f32x2*)(wi + q * 2);
      di[q] = pk_bf16(a.x, a.y);
      f32x2 b2 = *(const f32x2*)(wh + q * 2);
      dh[q] = pk_bf16(b2.x, b2.y);
    }
  }
  if (tid < GC) {
    int g = (tid >> 3) * H_ + j0 + (tid & 7);
    bsum[tid] = bi[g] + bh[g];
  }
  { // h0 = 0
    int b = tid >> 2; int c0 = (tid & 3) * 64;
    unsigned* d = &sH[b * SW + c0];
    for (int q = 0; q < 64; ++q) d[q] = 0u;
  }
  __syncthreads();

  const int lane = tid & 31;
  const int w    = tid >> 5;                 // wave 0..3
  const int kh   = lane >> 4;                // lane half
  const int lr   = lane & 15;
  const int m0   = (w & 1) * 16;             // batch tile
  const int n0   = (w >> 1) * 16;            // gate-col tile

  const float bias = bsum[n0 + lr];          // per-lane bias (same for all M)
  float creg0 = 0.0f, creg1 = 0.0f;          // cell state (2 items/thread)

  for (int t = 0; t < T_; ++t) {
    // ---- stage x_t (f32 -> bf16) ----------------------------------------
    {
      int b = tid >> 2; int c0 = (tid & 3) * 64;
      const float* src = x + ((size_t)b * T_ + t) * I_ + (size_t)c0 * 2;
      unsigned* d = &sX[b * SW + c0];
      for (int q = 0; q < 64; ++q) {
        f32x2 a = *(const f32x2*)(src + q * 2);
        d[q] = pk_bf16(a.x, a.y);
      }
    }
    // ---- stage h_{t-1} from global double buffer ------------------------
    if (t > 0) {
      const unsigned* src =
          (const unsigned*)hbuf + (((t - 1) & 1) ? (B_ * H_ / 2) : 0);
      int b = tid >> 2; int c0 = (tid & 3) * 64;
      const u32x4* s4 = (const u32x4*)(src + b * (H_ / 2) + c0);
      u32x4* d4 = (u32x4*)&sH[b * SW + c0];
      for (int q = 0; q < 16; ++q) d4[q] = s4[q];
    }
    __syncthreads();

    // ---- fused GEMM: gates = bias + x_t*Wi^T + h*Wh^T -------------------
    // Two independent WMMA accumulation chains interleaved for ILP.
    v8f acc0, acc1;
    #pragma unroll
    for (int r = 0; r < 8; ++r) { acc0[r] = bias; acc1[r] = 0.0f; }

    const int arow = (m0 + lr) * SW + kh * 8;        // A frag base (sX / sH)
    const int brow = (n0 + lr) * SW + kh * 8;        // B frag base (sWi / sWh)

    #pragma unroll 4
    for (int ks = 0; ks < 16; ++ks) {
      union { u32x4 u[2]; v16bf v; } Ax, Bx, Ah, Bh;
      const unsigned* axp = &sX [arow + ks * 16];
      const unsigned* ahp = &sH [arow + ks * 16];
      const unsigned* bxp = &sWi[brow + ks * 16];
      const unsigned* bhp = &sWh[brow + ks * 16];
      Ax.u[0] = *(const u32x4*)axp; Ax.u[1] = *(const u32x4*)(axp + 4);
      Bx.u[0] = *(const u32x4*)bxp; Bx.u[1] = *(const u32x4*)(bxp + 4);
      Ah.u[0] = *(const u32x4*)ahp; Ah.u[1] = *(const u32x4*)(ahp + 4);
      Bh.u[0] = *(const u32x4*)bhp; Bh.u[1] = *(const u32x4*)(bhp + 4);
      acc0 = __builtin_amdgcn_wmma_f32_16x16x32_bf16(
                 false, Ax.v, false, Bx.v, (short)0, acc0, false, false);
      acc1 = __builtin_amdgcn_wmma_f32_16x16x32_bf16(
                 false, Ah.v, false, Bh.v, (short)0, acc1, false, false);
    }
    #pragma unroll
    for (int r = 0; r < 8; ++r)              // C/D layout: VGPR r -> M=r(+8)
      gates[(m0 + r + kh * 8) * GSTR + n0 + lr] = acc0[r] + acc1[r];
    __syncthreads();

    // ---- activations + state update (2 adjacent-j items / thread) -------
    unsigned* hb32 = (unsigned*)(hbuf + ((t & 1) ? (B_ * H_) : 0));
    {
      int idx = tid * 2;                     // even
      int b = idx >> 3, jj = idx & 7;        // jj in {0,2,4,6}
      const float* gr = &gates[b * GSTR];
      float ig0 = gr[jj],      ig1 = gr[jj + 1];
      float fg0 = gr[8  + jj], fg1 = gr[9  + jj];
      float cg0 = gr[16 + jj], cg1 = gr[17 + jj];
      float og0 = gr[24 + jj], og1 = gr[25 + jj];
      float cn0 = sigm(fg0) * creg0 + sigm(ig0) * tanh_fast(cg0);
      float cn1 = sigm(fg1) * creg1 + sigm(ig1) * tanh_fast(cg1);
      float hn0 = sigm(og0) * tanh_fast(cn0);
      float hn1 = sigm(og1) * tanh_fast(cn1);
      creg0 = cn0; creg1 = cn1;
      f32x2 hv; hv.x = hn0; hv.y = hn1;
      *(f32x2*)(out + ((size_t)b * T_ + t) * H_ + j0 + jj) = hv;
      hb32[(b * H_ + j0 + jj) >> 1] = pk_bf16(hn0, hn1);
      if (t == T_ - 1) {
        f32x2 cv; cv.x = cn0; cv.y = cn1;
        *(f32x2*)(out + (size_t)B_ * T_ * H_ +
                  (size_t)b * H_ + j0 + jj) = hv;                 // h_fin
        *(f32x2*)(out + (size_t)B_ * T_ * H_ + (size_t)B_ * H_ +
                  (size_t)b * H_ + j0 + jj) = cv;                 // c_fin
      }
    }

    // ---- grid-wide barrier (arrive counter + generation) ----------------
    __syncthreads();
    if (tid == 0) {
      __threadfence();
      unsigned g = __atomic_load_n(&sync[1], __ATOMIC_ACQUIRE);
      unsigned a = __atomic_fetch_add(&sync[0], 1u, __ATOMIC_ACQ_REL);
      if (a == NWG - 1) {
        __atomic_store_n(&sync[0], 0u, __ATOMIC_RELAXED);
        __atomic_fetch_add(&sync[1], 1u, __ATOMIC_RELEASE);
      } else {
        while (__atomic_load_n(&sync[1], __ATOMIC_ACQUIRE) == g)
          __builtin_amdgcn_s_sleep(2);
      }
    }
    __syncthreads();
  }
}

extern "C" void kernel_launch(void* const* d_in, const int* in_sizes, int n_in,
                              void* d_out, int out_size, void* d_ws, size_t ws_size,
                              hipStream_t stream) {
  (void)in_sizes; (void)n_in; (void)out_size; (void)ws_size;
  const float* x  = (const float*)d_in[0];
  const float* Wi = (const float*)d_in[1];
  const float* bi = (const float*)d_in[2];
  const float* Wh = (const float*)d_in[3];
  const float* bh = (const float*)d_in[4];
  float* out = (float*)d_out;

  unsigned*       sync = (unsigned*)d_ws;                       // 2 counters
  unsigned short* hbuf = (unsigned short*)((char*)d_ws + 256);  // 2x[32][512] bf16

  lstm_init_kernel<<<1, 32, 0, stream>>>(sync);
  lstm_persist_kernel<<<NWG, NTHR, 0, stream>>>(x, Wi, bi, Wh, bh, out, sync, hbuf);
}
